// InActor_22531398434855
// MI455X (gfx1250) — matrix operand order repaired and checked by reading, not compiled, not run on hardware
//
#include <hip/hip_runtime.h>
#include <hip/hip_bf16.h>

typedef __attribute__((ext_vector_type(16))) _Float16 v16h;
typedef __attribute__((ext_vector_type(8)))  float    v8f;

#define NB_OBJ 5
#define NPERM  20

// ---------------- WMMA helpers (CDNA5 16x16x32 f16 -> f32) ----------------

static __device__ inline v8f wmma32(v16h a, v16h b, v8f c) {
  // D = A(16x32) * B(32x16) + C, f32 accumulate
  return __builtin_amdgcn_wmma_f32_16x16x32_f16(
      /*neg_a=*/false, a, /*neg_b=*/false, b,
      /*c_mod=*/(short)0, c, /*reuse_a=*/false, /*reuse_b=*/false);
}

// A fragment from row-major f16 LDS array: rows [mbase, mbase+16), cols [k0, k0+32)
// ISA layout: lanes 0-15 row M=lane, halves 0..7 = K 0..7, halves 8..15 = K 16..23
//             lanes 16-31 row M=lane-16, halves 0..7 = K 8..15, 8..15 = K 24..31
// -> two contiguous 16B chunks per lane => 2x ds_load_b128
static __device__ inline v16h load_a_frag(const _Float16* base, int stride,
                                          int mbase, int k0, int lane) {
  int r  = mbase + (lane & 15);
  int kb = k0 + ((lane >> 4) ? 8 : 0);
  const _Float16* p = base + (size_t)r * stride;
  v16h a;
#pragma unroll
  for (int i = 0; i < 8; ++i) {
    a[i]     = p[kb + i];
    a[8 + i] = p[kb + 16 + i];
  }
  return a;
}

// B fragment from pre-swizzled weights: each (ktile, ntile, lane) has its 16
// halves stored contiguously (32B, 32B-aligned) => 2x global_load_b128.
static __device__ inline v16h load_b_frag_sw(const _Float16* Wsw, int ntiles,
                                             int kt, int nt, int lane) {
  const v16h* p = (const v16h*)(Wsw + ((((size_t)kt * ntiles + nt) * 32 + lane) << 4));
  return *p;
}

// ---------------- weight prep: f32 -> f16, K zero-padded, B-fragment-major ----
// dst index: ((ktile*(N/16) + ntile)*32 + lane)*16 + h
//   k = ktile*32 + (lane>=16 ? 8 : 0) + (h&7) + (h>=8 ? 16 : 0) ; n = ntile*16 + (lane&15)

__global__ void k_cvt_swz(const float* __restrict__ src, _Float16* __restrict__ dst,
                          int K, int N, int Kpad) {
  int i = blockIdx.x * blockDim.x + threadIdx.x;
  int tot = Kpad * N;
  if (i >= tot) return;
  int h    = i & 15;
  int lane = (i >> 4) & 31;
  int tile = i >> 9;
  int ntiles = N >> 4;
  int nt = tile % ntiles;
  int kt = tile / ntiles;
  int k = kt * 32 + ((lane >> 4) ? 8 : 0) + (h & 7) + ((h >> 3) ? 16 : 0);
  int n = nt * 16 + (lane & 15);
  dst[i] = (k < K) ? (_Float16)src[(size_t)k * N + n] : (_Float16)0.0f;
}

// ---------------- kernel 1: edge MLP (8->256->128) + edge attention ----------------
// block = 4 batch elements -> 80 edge rows = 5 M-tiles; 8 waves.
// Hidden 256 streamed as 8 slabs of 32 cols so H never touches global memory.

__global__ __launch_bounds__(256) void k_mp_edge(
    const float* __restrict__ obs, const float* __restrict__ ag,
    const float* __restrict__ g,
    const _Float16* __restrict__ W1, const float* __restrict__ b1,
    const _Float16* __restrict__ W2, const float* __restrict__ b2,
    const float* __restrict__ eaW, const float* __restrict__ eab,
    const int* __restrict__ src, const int* __restrict__ dst,
    const int* __restrict__ inc, const int* __restrict__ pred,
    _Float16* __restrict__ phi_in) {
  __shared__ _Float16 sA[80 * 32];    // edge-MLP input, K padded 8->32
  __shared__ _Float16 sH[80 * 32];    // hidden slab (32 of 256 cols)
  __shared__ _Float16 sO[80 * 128];   // mp_out
  __shared__ float    sWgt[20 * 4];   // edge attention weights

  const int tid  = threadIdx.x;
  const int lane = tid & 31;
  // SGPR wave id -> scalar branches, EXEC stays all-ones around WMMA
  const int wave = __builtin_amdgcn_readfirstlane(tid) >> 5;
  const int b0   = blockIdx.x * 4;

  // ---- build input rows: [dg[pred0], dg[pred1], src.xyz, dst.xyz, 0...] ----
  for (int row = tid; row < 80; row += 256) {
    int bl = row / 20, e = row % 20;
    int b  = b0 + bl;
    const float* gb  = g   + (size_t)b * 40;
    const float* agb = ag  + (size_t)b * 40;
    const float* ob  = obs + (size_t)b * 85;
    int p0 = pred[2 * e], p1 = pred[2 * e + 1];
    int s = src[e], d = dst[e];
    _Float16* r = sA + row * 32;
    r[0] = (_Float16)(gb[p0] - agb[p0]);
    r[1] = (_Float16)(gb[p1] - agb[p1]);
#pragma unroll
    for (int c = 0; c < 3; ++c) {
      r[2 + c] = (_Float16)ob[10 + s * 15 + c];
      r[5 + c] = (_Float16)ob[10 + d * 15 + c];
    }
    for (int c = 8; c < 32; ++c) r[c] = (_Float16)0.0f;
  }
  __syncthreads();

  v8f c2[5];
#pragma unroll
  for (int m = 0; m < 5; ++m) { v8f z = {}; c2[m] = z; }

  const int col = lane & 15;
  const int rof = (lane >> 4) ? 8 : 0;

  for (int k2 = 0; k2 < 8; ++k2) {
    // layer 1: produce hidden cols [k2*32, k2*32+32) for all 80 rows
    // W1 swizzled: ktiles=1, ntiles=16
    for (int job = wave; job < 10; job += 8) {
      int m  = job % 5;
      int nn = job / 5;
      int nt = 2 * k2 + nn;
      v16h a  = load_a_frag(sA, 32, m * 16, 0, lane);
      v16h bf = load_b_frag_sw(W1, 16, 0, nt, lane);
      v8f  c  = {};
      c = wmma32(a, bf, c);
      float bias = b1[nt * 16 + col];
#pragma unroll
      for (int j = 0; j < 8; ++j) {
        float v = c[j] + bias;
        v = v > 0.0f ? v : 0.0f;
        sH[(m * 16 + j + rof) * 32 + nn * 16 + col] = (_Float16)v;
      }
    }
    __syncthreads();
    // layer 2: accumulate into registers; wave w owns output cols [w*16, w*16+16)
    // W2 swizzled: ktiles=8, ntiles=8; one B fragment feeds 5 WMMAs
    {
      v16h bf = load_b_frag_sw(W2, 8, k2, wave, lane);
#pragma unroll
      for (int m = 0; m < 5; ++m) {
        v16h a = load_a_frag(sH, 32, m * 16, 0, lane);
        c2[m] = wmma32(a, bf, c2[m]);
      }
    }
    __syncthreads();
  }

  // layer-2 epilogue: bias + relu -> sO
  {
    int n2 = wave * 16;
    float bias = b2[n2 + col];
#pragma unroll
    for (int m = 0; m < 5; ++m) {
#pragma unroll
      for (int j = 0; j < 8; ++j) {
        float v = c2[m][j] + bias;
        v = v > 0.0f ? v : 0.0f;
        sO[(m * 16 + j + rof) * 128 + n2 + col] = (_Float16)v;
      }
    }
  }
  __syncthreads();

  // ---- edge attention: softmax over the 4 incoming edges of each object ----
  if (tid < 20) {
    int bl = tid / 5, o = tid % 5;
    float lg[4];
    float mx = -1e30f;
#pragma unroll
    for (int e = 0; e < 4; ++e) {
      int edge = inc[o * 4 + e];
      const _Float16* rp = sO + (bl * 20 + edge) * 128;
      float dot = 0.0f;
      for (int c = 0; c < 128; ++c) dot += (float)rp[c] * eaW[c];
      lg[e] = dot + eab[0];
      mx = fmaxf(mx, lg[e]);
    }
    float sum = 0.0f;
#pragma unroll
    for (int e = 0; e < 4; ++e) { lg[e] = __expf(lg[e] - mx); sum += lg[e]; }
    float inv = 1.0f / sum;
#pragma unroll
    for (int e = 0; e < 4; ++e) sWgt[tid * 4 + e] = lg[e] * inv;
  }
  __syncthreads();

  // ---- write phi input rows: [body(10) | obj(15) | edge_feat(128) | pad(7)] ----
  for (int i = tid; i < 20 * 160; i += 256) {
    int row = i / 160, cc = i % 160;
    int bl = row / 5, o = row % 5;
    int b  = b0 + bl;
    float v;
    if (cc < 10) {
      v = obs[(size_t)b * 85 + cc];
    } else if (cc < 25) {
      v = obs[(size_t)b * 85 + 10 + o * 15 + (cc - 10)];
    } else if (cc < 153) {
      int c = cc - 25;
      v = 0.0f;
#pragma unroll
      for (int e = 0; e < 4; ++e) {
        int edge = inc[o * 4 + e];
        v += sWgt[row * 4 + e] * (float)sO[(bl * 20 + edge) * 128 + c];
      }
    } else {
      v = 0.0f;
    }
    phi_in[((size_t)(b * 5 + o)) * 160 + cc] = (_Float16)v;
  }
}

// ---------------- kernel 2: phi MLP 153(->160) -> 256 -> 128 ----------------

__global__ __launch_bounds__(256) void k_phi(
    const _Float16* __restrict__ phi_in,
    const _Float16* __restrict__ W1, const float* __restrict__ b1,
    const _Float16* __restrict__ W2, const float* __restrict__ b2,
    _Float16* __restrict__ phi_out) {
  __shared__ _Float16 sA[16 * 160];
  __shared__ _Float16 sH[16 * 256];
  const int tid = threadIdx.x, lane = tid & 31;
  const int wave = __builtin_amdgcn_readfirstlane(tid) >> 5;
  const size_t row0 = (size_t)blockIdx.x * 16;

  // 16 rows x 160 halves contiguous -> 1280 dwords
  {
    const unsigned int* sp = (const unsigned int*)(phi_in + row0 * 160);
    unsigned int* dp = (unsigned int*)sA;
    for (int i = tid; i < 1280; i += 256) dp[i] = sp[i];
  }
  __syncthreads();

  const int col = lane & 15;
  const int rof = (lane >> 4) ? 8 : 0;

  // layer 1: N=256; wave owns n-tiles {wave, wave+8}; K=160 (5 k-steps)
  // A fragment loaded once per k-step, feeds both n-tile accumulators.
  {
    v8f c0 = {}, c1 = {};
#pragma unroll
    for (int ks = 0; ks < 5; ++ks) {
      v16h a = load_a_frag(sA, 160, 0, ks * 32, lane);
      c0 = wmma32(a, load_b_frag_sw(W1, 16, ks, wave, lane), c0);
      c1 = wmma32(a, load_b_frag_sw(W1, 16, ks, wave + 8, lane), c1);
    }
    float bias0 = b1[wave * 16 + col];
    float bias1 = b1[(wave + 8) * 16 + col];
#pragma unroll
    for (int j = 0; j < 8; ++j) {
      float v0 = c0[j] + bias0;
      v0 = v0 > 0.0f ? v0 : 0.0f;
      sH[(j + rof) * 256 + wave * 16 + col] = (_Float16)v0;
      float v1 = c1[j] + bias1;
      v1 = v1 > 0.0f ? v1 : 0.0f;
      sH[(j + rof) * 256 + (wave + 8) * 16 + col] = (_Float16)v1;
    }
  }
  __syncthreads();

  // layer 2: N=128 (8 n-tiles, 1 per wave), K=256 (8 k-steps); W2 sw: ntiles=8
  {
    int n2 = wave * 16;
    v8f c = {};
#pragma unroll
    for (int ks = 0; ks < 8; ++ks) {
      v16h a  = load_a_frag(sH, 256, 0, ks * 32, lane);
      v16h bf = load_b_frag_sw(W2, 8, ks, wave, lane);
      c = wmma32(a, bf, c);
    }
    float bias = b2[n2 + col];
#pragma unroll
    for (int j = 0; j < 8; ++j) {
      float v = c[j] + bias;
      v = v > 0.0f ? v : 0.0f;
      phi_out[(row0 + j + rof) * 128 + n2 + col] = (_Float16)v;
    }
  }
}

// ---------------- kernel 3: softmax-attention pool over 5 objects ----------------

__global__ void k_pool(const _Float16* __restrict__ phi_out,
                       const float* __restrict__ saW, const float* __restrict__ sab,
                       _Float16* __restrict__ pooled, int B) {
  int b = blockIdx.x * blockDim.x + threadIdx.x;
  if (b >= B) return;
  const _Float16* base = phi_out + (size_t)b * 5 * 128;
  float lg[5];
  float mx = -1e30f;
#pragma unroll
  for (int o = 0; o < 5; ++o) {
    float d = 0.0f;
    for (int c = 0; c < 128; ++c) d += (float)base[o * 128 + c] * saW[c];
    lg[o] = d + sab[0];
    mx = fmaxf(mx, lg[o]);
  }
  float s = 0.0f;
#pragma unroll
  for (int o = 0; o < 5; ++o) { lg[o] = __expf(lg[o] - mx); s += lg[o]; }
  float inv = 1.0f / s;
  for (int c = 0; c < 128; ++c) {
    float v = 0.0f;
#pragma unroll
    for (int o = 0; o < 5; ++o) v += lg[o] * inv * (float)base[o * 128 + c];
    pooled[(size_t)b * 128 + c] = (_Float16)v;
  }
}

// ---------------- kernel 4: rho 128 -> 256 ----------------

__global__ __launch_bounds__(256) void k_rho(
    const _Float16* __restrict__ pooled, const _Float16* __restrict__ W,
    const float* __restrict__ bias, _Float16* __restrict__ r) {
  __shared__ _Float16 sA[16 * 128];
  const int tid = threadIdx.x, lane = tid & 31;
  const int wave = __builtin_amdgcn_readfirstlane(tid) >> 5;
  const size_t row0 = (size_t)blockIdx.x * 16;
  {
    const unsigned int* sp = (const unsigned int*)(pooled + row0 * 128);
    unsigned int* dp = (unsigned int*)sA;
    for (int i = tid; i < 1024; i += 256) dp[i] = sp[i];
  }
  __syncthreads();
  const int col = lane & 15;
  const int rof = (lane >> 4) ? 8 : 0;
  // N=256; wave owns n-tiles {wave, wave+8}; K=128 (4 k-steps); shared A frags
  {
    v8f c0 = {}, c1 = {};
#pragma unroll
    for (int ks = 0; ks < 4; ++ks) {
      v16h a = load_a_frag(sA, 128, 0, ks * 32, lane);
      c0 = wmma32(a, load_b_frag_sw(W, 16, ks, wave, lane), c0);
      c1 = wmma32(a, load_b_frag_sw(W, 16, ks, wave + 8, lane), c1);
    }
    float bias0 = bias[wave * 16 + col];
    float bias1 = bias[(wave + 8) * 16 + col];
#pragma unroll
    for (int j = 0; j < 8; ++j) {
      float v0 = c0[j] + bias0;
      v0 = v0 > 0.0f ? v0 : 0.0f;
      r[(row0 + j + rof) * 256 + wave * 16 + col] = (_Float16)v0;
      float v1 = c1[j] + bias1;
      v1 = v1 > 0.0f ? v1 : 0.0f;
      r[(row0 + j + rof) * 256 + (wave + 8) * 16 + col] = (_Float16)v1;
    }
  }
}

// ---------------- kernel 5: heads (mean + clipped logstd) ----------------

__global__ void k_heads(const _Float16* __restrict__ r,
                        const float* __restrict__ mW, const float* __restrict__ mb,
                        const float* __restrict__ lW, const float* __restrict__ lb,
                        float* __restrict__ out, int B) {
  int i = blockIdx.x * blockDim.x + threadIdx.x;
  if (i >= B * 8) return;
  int b = i >> 3, j = i & 7;
  const _Float16* rp = r + (size_t)b * 256;
  if (j < 4) {
    float v = mb[j];
    for (int c = 0; c < 256; ++c) v += (float)rp[c] * mW[c * 4 + j];
    out[(size_t)b * 4 + j] = v;
  } else {
    int jj = j - 4;
    float v = lb[jj];
    for (int c = 0; c < 256; ++c) v += (float)rp[c] * lW[c * 4 + jj];
    v = fminf(fmaxf(v, -20.0f), 2.0f);
    out[(size_t)B * 4 + (size_t)b * 4 + jj] = v;
  }
}

// ---------------- launcher ----------------

extern "C" void kernel_launch(void* const* d_in, const int* in_sizes, int n_in,
                              void* d_out, int out_size, void* d_ws, size_t ws_size,
                              hipStream_t stream) {
  const float* obs    = (const float*)d_in[0];
  const float* ag     = (const float*)d_in[1];
  const float* g      = (const float*)d_in[2];
  const float* mp_W1  = (const float*)d_in[3];
  const float* mp_b1  = (const float*)d_in[4];
  const float* mp_W2  = (const float*)d_in[5];
  const float* mp_b2  = (const float*)d_in[6];
  const float* ea_W   = (const float*)d_in[7];
  const float* ea_b   = (const float*)d_in[8];
  const float* phi_W1 = (const float*)d_in[9];
  const float* phi_b1 = (const float*)d_in[10];
  const float* phi_W2 = (const float*)d_in[11];
  const float* phi_b2 = (const float*)d_in[12];
  const float* sa_W   = (const float*)d_in[13];
  const float* sa_b   = (const float*)d_in[14];
  const float* rho_W1 = (const float*)d_in[15];
  const float* rho_b1 = (const float*)d_in[16];
  const float* mean_W = (const float*)d_in[17];
  const float* mean_b = (const float*)d_in[18];
  const float* ls_W   = (const float*)d_in[19];
  const float* ls_b   = (const float*)d_in[20];
  const int*   src    = (const int*)d_in[21];
  const int*   dst    = (const int*)d_in[22];
  const int*   inc    = (const int*)d_in[23];
  const int*   pred   = (const int*)d_in[24];
  float* out = (float*)d_out;

  const int B = in_sizes[0] / 85;

  // workspace carve-up (256B aligned)
  char* ws = (char*)d_ws;
  size_t off = 0;
  auto take = [&](size_t halves) -> _Float16* {
    _Float16* p = (_Float16*)(ws + off);
    off += ((halves * sizeof(_Float16) + 255) / 256) * 256;
    return p;
  };
  _Float16* w1f   = take(32 * 256);
  _Float16* w2f   = take(256 * 128);
  _Float16* pw1f  = take(160 * 256);
  _Float16* pw2f  = take(256 * 128);
  _Float16* rw1f  = take(128 * 256);
  _Float16* phiin = take((size_t)B * 5 * 160);
  _Float16* phio  = take((size_t)B * 5 * 128);
  _Float16* pool  = take((size_t)B * 128);
  _Float16* rbuf  = take((size_t)B * 256);

  // weight prep: f16 convert + zero-pad K + swizzle into B-fragment layout
  k_cvt_swz<<<(32 * 256 + 255) / 256, 256, 0, stream>>>(mp_W1, w1f, 8, 256, 32);
  k_cvt_swz<<<(256 * 128 + 255) / 256, 256, 0, stream>>>(mp_W2, w2f, 256, 128, 256);
  k_cvt_swz<<<(160 * 256 + 255) / 256, 256, 0, stream>>>(phi_W1, pw1f, 153, 256, 160);
  k_cvt_swz<<<(256 * 128 + 255) / 256, 256, 0, stream>>>(phi_W2, pw2f, 256, 128, 256);
  k_cvt_swz<<<(128 * 256 + 255) / 256, 256, 0, stream>>>(rho_W1, rw1f, 128, 256, 128);

  // 1) edge MLP + edge attention -> phi input
  k_mp_edge<<<B / 4, 256, 0, stream>>>(obs, ag, g, w1f, mp_b1, w2f, mp_b2,
                                       ea_W, ea_b, src, dst, inc, pred, phiin);
  // 2) phi MLP
  k_phi<<<(B * 5) / 16, 256, 0, stream>>>(phiin, pw1f, phi_b1, pw2f, phi_b2, phio);
  // 3) attention pool
  k_pool<<<(B + 255) / 256, 256, 0, stream>>>(phio, sa_W, sa_b, pool, B);
  // 4) rho
  k_rho<<<B / 16, 256, 0, stream>>>(pool, rw1f, rho_b1, rbuf);
  // 5) heads
  k_heads<<<(B * 8 + 255) / 256, 256, 0, stream>>>(rbuf, mean_W, mean_b, ls_W, ls_b, out, B);
}
